// DeepRicci_42752104464582
// MI455X (gfx1250) — matrix operand order repaired and checked by reading, not compiled, not run on hardware
//
#include <hip/hip_runtime.h>
#include <math.h>

#define NN     4096
#define DIN    512
#define DHYLA  256
#define DRIE   32
#define DHID   512
#define DX     768    // 3*DHYLA
#define TOPK_  30

typedef __bf16 bf16;
typedef __bf16 v16bf __attribute__((ext_vector_type(16)));
typedef float  v8f   __attribute__((ext_vector_type(8)));

// ---------------------------------------------------------------------------
// WMMA GEMM: C(f32, MxN) = A(bf16, MxK, row-major, lda) * Bt^T
//   Bt stored row-major NxK (ldb = row stride)  ->  contiguous per-lane loads
// One wave computes a 32x32 tile (2x2 of 16x16, 4 accumulators, A/B reuse x2).
// 4 waves per 128-thread block.  M,N multiples of 32; K multiple of 32.
// EXEC stays all-ones (WMMA requirement).
// ---------------------------------------------------------------------------
__global__ void gemm32_wmma_bf16(const bf16* __restrict__ A, const bf16* __restrict__ Bt,
                                 float* __restrict__ C, int M, int N, int K,
                                 int lda, int ldb, int ldc)
{
    (void)M;
    const int lane = threadIdx.x & 31;
    const int warp = threadIdx.x >> 5;
    const int tn32 = N >> 5;
    const int tile = blockIdx.x * 4 + warp;
    const int tm   = (tile / tn32) << 5;
    const int tn   = (tile % tn32) << 5;
    const int hf   = lane >> 4;   // half-wave select
    const int lr   = lane & 15;

    v8f acc00 = {0.f,0.f,0.f,0.f,0.f,0.f,0.f,0.f};
    v8f acc01 = acc00, acc10 = acc00, acc11 = acc00;

    // A-matrix 16-bit layout: lane lr holds row; K = k0 + hf*8 + {e | e+16-8}
    const bf16* ap0 = A  + (size_t)(tm + lr) * lda + hf * 8;
    const bf16* ap1 = ap0 + (size_t)16 * lda;
    // B-matrix 16-bit layout: lane lr holds column; K = k0 + hf*8 + {e | e+16-8}
    const bf16* bp0 = Bt + (size_t)(tn + lr) * ldb + hf * 8;
    const bf16* bp1 = bp0 + (size_t)16 * ldb;

    for (int k0 = 0; k0 < K; k0 += 32) {
        v16bf a0, a1, b0, b1;
#pragma unroll
        for (int e = 0; e < 8; ++e) {
            a0[e]     = ap0[k0 + e];
            a0[e + 8] = ap0[k0 + 16 + e];
            a1[e]     = ap1[k0 + e];
            a1[e + 8] = ap1[k0 + 16 + e];
        }
#pragma unroll
        for (int e = 0; e < 8; ++e) { b0[e] = bp0[k0 + e]; b0[e + 8] = bp0[k0 + 16 + e]; b1[e] = bp1[k0 + e]; b1[e + 8] = bp1[k0 + 16 + e]; }

        // cover next K-block latency (global_prefetch_b8; OOB prefetch is dropped)
        __builtin_prefetch((const void*)(ap0 + k0 + 128), 0, 1);
        __builtin_prefetch((const void*)(ap1 + k0 + 128), 0, 1);
        __builtin_prefetch((const void*)(bp0 + k0 + 128), 0, 1);
        __builtin_prefetch((const void*)(bp1 + k0 + 128), 0, 1);

        acc00 = __builtin_amdgcn_wmma_f32_16x16x32_bf16(false, a0, false, b0, (short)0, acc00, false, false);
        acc01 = __builtin_amdgcn_wmma_f32_16x16x32_bf16(false, a0, false, b1, (short)0, acc01, false, false);
        acc10 = __builtin_amdgcn_wmma_f32_16x16x32_bf16(false, a1, false, b0, (short)0, acc10, false, false);
        acc11 = __builtin_amdgcn_wmma_f32_16x16x32_bf16(false, a1, false, b1, (short)0, acc11, false, false);
    }
    // C/D layout: VGPR r -> row + r + 8*hf, col + lr
    float* c00 = C + (size_t)(tm + hf * 8) * ldc + tn + lr;
    float* c10 = c00 + (size_t)16 * ldc;
#pragma unroll
    for (int r = 0; r < 8; ++r) {
        c00[(size_t)r * ldc]      = acc00[r];
        c00[(size_t)r * ldc + 16] = acc01[r];
        c10[(size_t)r * ldc]      = acc10[r];
        c10[(size_t)r * ldc + 16] = acc11[r];
    }
}

// tiled transpose with bf16 convert: out[C x R] = (bf16) in[R x C]^T
// grid(C/32, R/32), block(32, 8); R,C multiples of 32
__global__ void k_transpose_f32_bf16(bf16* __restrict__ out, const float* __restrict__ in,
                                     int R, int C) {
    __shared__ float tile[32][33];
    const int c0 = blockIdx.x * 32;
    const int r0 = blockIdx.y * 32;
    const int tx = threadIdx.x, ty = threadIdx.y;
#pragma unroll
    for (int i = 0; i < 32; i += 8)
        tile[ty + i][tx] = in[(size_t)(r0 + ty + i) * C + c0 + tx];
    __syncthreads();
#pragma unroll
    for (int i = 0; i < 32; i += 8)
        out[(size_t)(c0 + ty + i) * R + r0 + tx] = (bf16)tile[tx][ty + i];
}

// --------------------------- elementwise helpers ---------------------------
__global__ void k_zero(float* p, int n) {
    int i = blockIdx.x * blockDim.x + threadIdx.x;
    if (i < n) p[i] = 0.0f;
}

__global__ void k_f32_to_bf16(bf16* dst, const float* src, int n) {
    int i = blockIdx.x * blockDim.x + threadIdx.x;
    if (i < n) dst[i] = (bf16)src[i];
}

// normalize rows of src (rows x cols) and convert to bf16
__global__ void k_rownorm_bf16(bf16* dst, const float* src, int rows, int cols) {
    int r = blockIdx.x * blockDim.x + threadIdx.x;
    if (r >= rows) return;
    float s = 0.0f;
    for (int c = 0; c < cols; ++c) { float v = src[r * cols + c]; s += v * v; }
    float inv = rsqrtf(fmaxf(s, 1e-30f));
    for (int c = 0; c < cols; ++c) dst[r * cols + c] = (bf16)(src[r * cols + c] * inv);
}

__global__ void k_rowsumsq(float* out, const float* src, int rows, int cols) {
    int r = blockIdx.x * blockDim.x + threadIdx.x;
    if (r >= rows) return;
    float s = 0.0f;
    for (int c = 0; c < cols; ++c) { float v = src[r * cols + c]; s += v * v; }
    out[r] = s;
}

// X[r,c] = scale * cos(X[r,c] + bias[c])   on a rows x cols sub-block, stride ld
__global__ void k_cos_ep(float* X, int rows, int cols, int ld, const float* bias, float scale) {
    int i = blockIdx.x * blockDim.x + threadIdx.x;
    if (i >= rows * cols) return;
    int r = i / cols, c = i % cols;
    float* p = X + (size_t)r * ld + c;
    *p = scale * cosf(*p + bias[c]);
}

// hyperbolic/spherical laplacian feature epilogue (n=32 -> (n-1)/2 = 15.5)
__global__ void k_hyla_ep(float* X, int rows, int cols, int ld, const float* ssum,
                          const float* eig, const float* bias, int kneg) {
    int i = blockIdx.x * blockDim.x + threadIdx.x;
    if (i >= rows * cols) return;
    int r = i / cols, c = i % cols;
    float* p = X + (size_t)r * ld + c;
    float zb = *p;
    float dist = (zb - ssum[r]) / (1.0f - zb);
    float P = kneg ? atanhf(dist) : atanf(dist);
    *p = expf(15.5f * P) * cosf(eig[c] * P + bias[c]);
}

__global__ void k_bias_ep(float* X, int rows, int cols, int ld, const float* bias, int relu) {
    int i = blockIdx.x * blockDim.x + threadIdx.x;
    if (i >= rows * cols) return;
    int r = i / cols, c = i % cols;
    float* p = X + (size_t)r * ld + c;
    float v = *p + bias[c];
    *p = relu ? fmaxf(v, 0.0f) : v;
}

// f[i] = sum_d x[i,d]*w[d] + b[0]
__global__ void k_lip(float* f, const float* x, const float* w, const float* b, int rows, int cols) {
    int i = blockIdx.x * blockDim.x + threadIdx.x;
    if (i >= rows) return;
    float s = 0.0f;
    for (int d = 0; d < cols; ++d) s += x[(size_t)i * cols + d] * w[d];
    f[i] = s + b[0];
}

// row softmax with input pre-scale (score/H)
__global__ void k_softmax_row(float* S, int n, float scale) {
    __shared__ float red[256];
    const int row = blockIdx.x, tid = threadIdx.x;
    float m = -1e30f;
    for (int j = tid; j < n; j += 256) m = fmaxf(m, S[(size_t)row * n + j] * scale);
    red[tid] = m; __syncthreads();
    for (int s = 128; s > 0; s >>= 1) { if (tid < s) red[tid] = fmaxf(red[tid], red[tid + s]); __syncthreads(); }
    m = red[0]; __syncthreads();
    float sum = 0.0f;
    for (int j = tid; j < n; j += 256) {
        float e = expf(S[(size_t)row * n + j] * scale - m);
        S[(size_t)row * n + j] = e; sum += e;
    }
    red[tid] = sum; __syncthreads();
    for (int s = 128; s > 0; s >>= 1) { if (tid < s) red[tid] += red[tid + s]; __syncthreads(); }
    float inv = 1.0f / red[0];
    for (int j = tid; j < n; j += 256) S[(size_t)row * n + j] *= inv;
}

// K-th largest value per row via repeated masked max
__global__ void k_kth_row(const float* S, int n, int K, float* kth) {
    __shared__ float red[256];
    const int row = blockIdx.x, tid = threadIdx.x;
    float pv = 1e30f;
    for (int it = 0; it < K; ++it) {
        float m = -1e30f;
        for (int j = tid; j < n; j += 256) {
            float v = S[(size_t)row * n + j];
            if (v < pv) m = fmaxf(m, v);
        }
        red[tid] = m; __syncthreads();
        for (int s = 128; s > 0; s >>= 1) { if (tid < s) red[tid] = fmaxf(red[tid], red[tid + s]); __syncthreads(); }
        pv = red[0]; __syncthreads();
    }
    if (tid == 0) kth[row] = pv;
}

// lA = relu(0.5*(thresh(S) + thresh(S)^T)) in place; pair (i,j) handled once
__global__ void k_thresh_sym(float* S, const float* kth, int n) {
    int idx = blockIdx.x * blockDim.x + threadIdx.x;
    if (idx >= n * n) return;
    int i = idx / n, j = idx % n;
    if (i > j) return;
    float a = S[(size_t)i * n + j]; a = (a >= kth[i]) ? a : 0.0f;
    float b = S[(size_t)j * n + i]; b = (b >= kth[j]) ? b : 0.0f;
    float v = fmaxf(0.5f * (a + b), 0.0f);
    S[(size_t)i * n + j] = v;
    S[(size_t)j * n + i] = v;
}

__global__ void k_rowsum(const float* S, float* out, int n) {
    __shared__ float red[256];
    const int row = blockIdx.x, tid = threadIdx.x;
    float s = 0.0f;
    for (int j = tid; j < n; j += 256) s += S[(size_t)row * n + j];
    red[tid] = s; __syncthreads();
    for (int st = 128; st > 0; st >>= 1) { if (tid < st) red[tid] += red[tid + st]; __syncthreads(); }
    if (tid == 0) out[row] = red[0];
}

__global__ void k_colsum(const float* S, float* out, int n) {
    __shared__ float red[256];
    const int col = blockIdx.x, tid = threadIdx.x;
    float s = 0.0f;
    for (int r = tid; r < n; r += 256) s += S[(size_t)r * n + col];
    red[tid] = s; __syncthreads();
    for (int st = 128; st > 0; st >>= 1) { if (tid < st) red[tid] += red[tid + st]; __syncthreads(); }
    if (tid == 0) out[col] = red[0];
}

__global__ void k_deginv(const float* deg, float* dsym, float* drow, int n) {
    int i = blockIdx.x * blockDim.x + threadIdx.x;
    if (i >= n) return;
    float d = deg[i];
    dsym[i] = (d > 0.0f) ? rsqrtf(d) : 0.0f;
    drow[i] = (d > 0.0f) ? 1.0f / d : 0.0f;
}

// A_sym = (1-r)*norm_sym(lA) + r*norm_sym(A); also produce bf16 copy
__global__ void k_mix_adj(float* outA, bf16* outAbf, const float* lA, const float* A,
                          const float* dls, const float* das, const float* ratio, int n) {
    int idx = blockIdx.x * blockDim.x + threadIdx.x;
    if (idx >= n * n) return;
    int i = idx / n, j = idx % n;
    float r = *ratio;
    float v = (1.0f - r) * lA[idx] * dls[i] * dls[j] + r * A[idx] * das[i] * das[j];
    outA[idx]  = v;
    outAbf[idx] = (bf16)v;
}

// top-K indices per row (argmax iterations, tie -> lower index first)
__global__ void k_topk_idx(const float* M, int n, int K, int* tgt) {
    __shared__ float sv[256];
    __shared__ int   si[256];
    const int row = blockIdx.x, tid = threadIdx.x;
    float pv = 1e30f;
    int   pidx = -1;
    for (int it = 0; it < K; ++it) {
        float bv = -1e30f; int bi = 0x7fffffff;
        for (int j = tid; j < n; j += 256) {
            float v = M[(size_t)row * n + j];
            bool elig = (v < pv) || (v == pv && j > pidx);
            if (elig && (v > bv || (v == bv && j < bi))) { bv = v; bi = j; }
        }
        sv[tid] = bv; si[tid] = bi; __syncthreads();
        for (int s = 128; s > 0; s >>= 1) {
            if (tid < s) {
                if (sv[tid + s] > sv[tid] || (sv[tid + s] == sv[tid] && si[tid + s] < si[tid])) {
                    sv[tid] = sv[tid + s]; si[tid] = si[tid + s];
                }
            }
            __syncthreads();
        }
        pv = sv[0]; pidx = si[0]; __syncthreads();
        if (tid == 0) tgt[row * K + it] = pidx;
    }
}

// Ricci structure loss accumulation (ALPHA=0.5 -> c=-1; R_P=2, S_P=1)
__global__ void k_structure(const int* tgt, const float* lA, const float* A, const float* f,
                            const float* dlr, const float* dar, const float* ratio,
                            float* accum, int n, int K) {
    int t = blockIdx.x * blockDim.x + threadIdx.x;
    if (t >= n * K) return;
    int i = t / K;
    int j = tgt[t];
    float r = *ratio;
    float ar = (1.0f - r) * lA[(size_t)i * n + j] * dlr[i] + r * A[(size_t)i * n + j] * dar[i];
    const float alpha = 0.5f;
    const float c = 1.0f - 1.0f / alpha;   // = -1
    float lw = ((j == i) ? 1.0f : 0.0f) - c * ar;
    float w  = fmaxf(alpha * lw * (f[i] - f[j]), 0.0f);
    float pr = 1.0f / (1.0f + expf(-(2.0f - w) / 1.0f));
    atomicAdd(&accum[2], -logf(pr + 1e-5f));
}

// build x1=[zE, feat_h], x2=[zE, feat_s] in bf16 + row norms
__global__ void k_x12(const float* zE, const float* x, bf16* x1, bf16* x2,
                      float* n1, float* n2, int n) {
    int i = blockIdx.x * blockDim.x + threadIdx.x;
    if (i >= n) return;
    const float* z  = zE + (size_t)i * DHYLA;
    const float* fh = x + (size_t)i * DX + DHYLA;
    const float* fs = x + (size_t)i * DX + 2 * DHYLA;
    float s1 = 0.0f, s2 = 0.0f;
    for (int c = 0; c < DHYLA; ++c) {
        float v = z[c]; s1 += v * v; s2 += v * v;
        x1[(size_t)i * 512 + c] = (bf16)v;
        x2[(size_t)i * 512 + c] = (bf16)v;
    }
    for (int c = 0; c < DHYLA; ++c) {
        float a = fh[c]; s1 += a * a; x1[(size_t)i * 512 + DHYLA + c] = (bf16)a;
        float b = fs[c]; s2 += b * b; x2[(size_t)i * 512 + DHYLA + c] = (bf16)b;
    }
    n1[i] = sqrtf(s1);
    n2[i] = sqrtf(s2);
}

// sim = exp(dot / (n1[i]*n2[j]) / TEMP), TEMP=0.1
__global__ void k_sim_ep(float* S, const float* n1, const float* n2, int n) {
    int idx = blockIdx.x * blockDim.x + threadIdx.x;
    if (idx >= n * n) return;
    int i = idx / n, j = idx % n;
    S[idx] = expf(S[idx] / (n1[i] * n2[j]) * 10.0f);
}

__global__ void k_diag(const float* S, float* d, int n) {
    int i = blockIdx.x * blockDim.x + threadIdx.x;
    if (i < n) d[i] = S[(size_t)i * n + i];
}

__global__ void k_info(const float* rowsum, const float* colsum, const float* diag,
                       float* accum, int n) {
    __shared__ float r1[256], r2[256];
    const int tid = threadIdx.x;
    float a = 0.0f, b = 0.0f;
    for (int j = tid; j < n; j += 256) {
        float p = diag[j];
        a += -logf(p / (colsum[j] - p));
        b += -logf(p / (rowsum[j] - p));
    }
    r1[tid] = a; r2[tid] = b; __syncthreads();
    for (int s = 128; s > 0; s >>= 1) {
        if (tid < s) { r1[tid] += r1[tid + s]; r2[tid] += r2[tid + s]; }
        __syncthreads();
    }
    if (tid == 0) { accum[0] = r1[0]; accum[1] = r2[0]; }
}

__global__ void k_final(const float* accum, float* loss_out, int n, int K) {
    if (blockIdx.x == 0 && threadIdx.x == 0) {
        float l1 = accum[0] / (float)n;
        float l2 = accum[1] / (float)n;
        float info = 0.5f * (l1 + l2);
        float st = accum[2] / (float)(n * K);
        *loss_out = info + 1.0f * st;   // GAMMA = 1
    }
}

__global__ void k_copy_pz(float* pz, const float* zE, const float* eh, const float* es, int n) {
    int idx = blockIdx.x * blockDim.x + threadIdx.x;
    if (idx >= n * 320) return;
    int i = idx / 320, c = idx % 320;
    float v;
    if (c < 256)      v = zE[(size_t)i * 256 + c];
    else if (c < 288) v = eh[(size_t)i * 32 + (c - 256)];
    else              v = es[(size_t)i * 32 + (c - 288)];
    pz[idx] = v;
}

__global__ void k_copy_nf(float* nf, const float* feat, const float* eh, const float* es, int n) {
    int idx = blockIdx.x * blockDim.x + threadIdx.x;
    if (idx >= n * 576) return;
    int i = idx / 576, c = idx % 576;
    float v;
    if (c < 512)      v = feat[(size_t)i * 512 + c];
    else if (c < 544) v = eh[(size_t)i * 32 + (c - 512)];
    else              v = es[(size_t)i * 32 + (c - 544)];
    nf[idx] = v;
}

// ---------------------------------------------------------------------------
extern "C" void kernel_launch(void* const* d_in, const int* in_sizes, int n_in,
                              void* d_out, int out_size, void* d_ws, size_t ws_size,
                              hipStream_t stream)
{
    (void)in_sizes; (void)n_in; (void)out_size; (void)ws_size;

    const float* feature  = (const float*)d_in[0];
    const float* A        = (const float*)d_in[1];
    const float* init_adj = (const float*)d_in[2];
    const float* emb_h    = (const float*)d_in[3];
    const float* emb_s    = (const float*)d_in[4];
    const float* b_eu     = (const float*)d_in[5];
    const float* bias_eu  = (const float*)d_in[6];
    const float* b_h      = (const float*)d_in[7];
    const float* eig_h    = (const float*)d_in[8];
    const float* bias_h   = (const float*)d_in[9];
    const float* b_s      = (const float*)d_in[10];
    const float* eig_s    = (const float*)d_in[11];
    const float* bias_s   = (const float*)d_in[12];
    const float* wq_w     = (const float*)d_in[13];
    const float* wq_b     = (const float*)d_in[14];
    const float* lip_w    = (const float*)d_in[15];
    const float* lip_b    = (const float*)d_in[16];
    const float* enc_w0   = (const float*)d_in[17];
    const float* enc_b0   = (const float*)d_in[18];
    const float* enc_w1   = (const float*)d_in[19];
    const float* enc_b1   = (const float*)d_in[20];
    const float* ratio    = (const float*)d_in[21];

    float* out_pz   = (float*)d_out;                 // [NN,320]
    float* out_nf   = out_pz + (size_t)NN * 320;     // [NN,576]
    float* out_Asym = out_nf + (size_t)NN * 576;     // [NN,NN]
    float* out_loss = out_Asym + (size_t)NN * NN;    // [1]

    size_t off = 0;
    char* base = (char*)d_ws;
    auto alloc = [&](size_t bytes) -> char* {
        char* p = base + off;
        off += (bytes + 255) & ~(size_t)255;
        return p;
    };
    float* x       = (float*)alloc((size_t)NN * DX * 4);
    bf16*  x_bf    = (bf16*) alloc((size_t)NN * DX * 2);
    float* S       = (float*)alloc((size_t)NN * NN * 4);     // score -> lA -> sim
    float* q       = (float*)alloc((size_t)NN * DHYLA * 4);
    bf16*  q_bf    = (bf16*) alloc((size_t)NN * DHYLA * 2);
    bf16*  feat_bf = (bf16*) alloc((size_t)NN * DIN * 2);
    bf16*  beu_bf  = (bf16*) alloc((size_t)DHYLA * DIN * 2);   // already NxK
    bf16*  bh_bf   = (bf16*) alloc((size_t)DHYLA * DRIE * 2);  // already NxK
    bf16*  bs_bf   = (bf16*) alloc((size_t)DHYLA * DRIE * 2);
    bf16*  eh_bf   = (bf16*) alloc((size_t)NN * DRIE * 2);
    bf16*  es_bf   = (bf16*) alloc((size_t)NN * DRIE * 2);
    float* ssum_h  = (float*)alloc(NN * 4);
    float* ssum_s  = (float*)alloc(NN * 4);
    bf16*  wqt     = (bf16*) alloc((size_t)DHYLA * DX * 2);    // wq_w^T  [256,768]
    bf16*  w0t     = (bf16*) alloc((size_t)DHID * DIN * 2);    // enc_w0^T [512,512]
    bf16*  w1t     = (bf16*) alloc((size_t)DHYLA * DHID * 2);  // enc_w1^T [256,512]
    float* XW      = (float*)alloc((size_t)NN * DHID * 4);
    bf16*  XWt     = (bf16*) alloc((size_t)DHID * NN * 2);     // XW^T [512,4096]
    bf16*  Asym_bf = (bf16*) alloc((size_t)NN * NN * 2);
    float* hbuf    = (float*)alloc((size_t)NN * DHID * 4);
    bf16*  h_bf    = (bf16*) alloc((size_t)NN * DHID * 2);
    float* HW1     = (float*)alloc((size_t)NN * DHYLA * 4);
    bf16*  HW1t    = (bf16*) alloc((size_t)DHYLA * NN * 2);    // HW1^T [256,4096]
    float* zE      = (float*)alloc((size_t)NN * DHYLA * 4);
    bf16*  x1_bf   = (bf16*) alloc((size_t)NN * 512 * 2);
    bf16*  x2_bf   = (bf16*) alloc((size_t)NN * 512 * 2);
    float* nrm1    = (float*)alloc(NN * 4);
    float* nrm2    = (float*)alloc(NN * 4);
    float* rowsum  = (float*)alloc(NN * 4);
    float* colsum  = (float*)alloc(NN * 4);
    float* diagv   = (float*)alloc(NN * 4);
    float* deg_l   = (float*)alloc(NN * 4);
    float* deg_a   = (float*)alloc(NN * 4);
    float* dls     = (float*)alloc(NN * 4);
    float* dlr     = (float*)alloc(NN * 4);
    float* das     = (float*)alloc(NN * 4);
    float* dar     = (float*)alloc(NN * 4);
    float* kth     = (float*)alloc(NN * 4);
    float* fvec    = (float*)alloc(NN * 4);
    int*   tgt     = (int*)  alloc((size_t)NN * TOPK_ * 4);
    float* accum   = (float*)alloc(4 * 4);

    int n;
    k_zero<<<1, 32, 0, stream>>>(accum, 4);

    // ---- bf16 conversions / weight prep (transpose all KxN weights to NxK) ----
    n = NN * DIN;      k_f32_to_bf16<<<(n + 255) / 256, 256, 0, stream>>>(feat_bf, feature, n);
    n = DHYLA * DIN;   k_f32_to_bf16<<<(n + 255) / 256, 256, 0, stream>>>(beu_bf, b_eu, n);
    k_rownorm_bf16<<<1, 256, 0, stream>>>(bh_bf, b_h, DHYLA, DRIE);
    k_rownorm_bf16<<<1, 256, 0, stream>>>(bs_bf, b_s, DHYLA, DRIE);
    n = NN * DRIE;     k_f32_to_bf16<<<(n + 255) / 256, 256, 0, stream>>>(eh_bf, emb_h, n);
    n = NN * DRIE;     k_f32_to_bf16<<<(n + 255) / 256, 256, 0, stream>>>(es_bf, emb_s, n);
    k_rowsumsq<<<(NN + 255) / 256, 256, 0, stream>>>(ssum_h, emb_h, NN, DRIE);
    k_rowsumsq<<<(NN + 255) / 256, 256, 0, stream>>>(ssum_s, emb_s, NN, DRIE);
    k_transpose_f32_bf16<<<dim3(DHYLA / 32, DX / 32),   dim3(32, 8), 0, stream>>>(wqt, wq_w,   DX,   DHYLA);
    k_transpose_f32_bf16<<<dim3(DHID / 32,  DIN / 32),  dim3(32, 8), 0, stream>>>(w0t, enc_w0, DIN,  DHID);
    k_transpose_f32_bf16<<<dim3(DHYLA / 32, DHID / 32), dim3(32, 8), 0, stream>>>(w1t, enc_w1, DHID, DHYLA);

    // ---- features x = [feat_eu | feat_h | feat_s] ----
    gemm32_wmma_bf16<<<256, 128, 0, stream>>>(feat_bf, beu_bf, x,          NN, DHYLA, DIN,  DIN,  DIN,  DX);
    n = NN * DHYLA;    k_cos_ep<<<(n + 255) / 256, 256, 0, stream>>>(x, NN, DHYLA, DX, bias_eu, 1.41421356f);
    gemm32_wmma_bf16<<<256, 128, 0, stream>>>(eh_bf, bh_bf, x + DHYLA,     NN, DHYLA, DRIE, DRIE, DRIE, DX);
    k_hyla_ep<<<(n + 255) / 256, 256, 0, stream>>>(x + DHYLA, NN, DHYLA, DX, ssum_h, eig_h, bias_h, 1);
    gemm32_wmma_bf16<<<256, 128, 0, stream>>>(es_bf, bs_bf, x + 2 * DHYLA, NN, DHYLA, DRIE, DRIE, DRIE, DX);
    k_hyla_ep<<<(n + 255) / 256, 256, 0, stream>>>(x + 2 * DHYLA, NN, DHYLA, DX, ssum_s, eig_s, bias_s, 0);

    // ---- q = x @ wq_w + wq_b ; f = x @ lip_w + lip_b ----
    n = NN * DX;       k_f32_to_bf16<<<(n + 255) / 256, 256, 0, stream>>>(x_bf, x, n);
    gemm32_wmma_bf16<<<256, 128, 0, stream>>>(x_bf, wqt, q, NN, DHYLA, DX, DX, DX, DHYLA);
    n = NN * DHYLA;    k_bias_ep<<<(n + 255) / 256, 256, 0, stream>>>(q, NN, DHYLA, DHYLA, wq_b, 0);
    k_f32_to_bf16<<<(n + 255) / 256, 256, 0, stream>>>(q_bf, q, n);
    k_lip<<<(NN + 255) / 256, 256, 0, stream>>>(fvec, x, lip_w, lip_b, NN, DX);

    // ---- score = softmax((q q^T)/H) ; top-k threshold ; symmetrize ----
    gemm32_wmma_bf16<<<4096, 128, 0, stream>>>(q_bf, q_bf, S, NN, NN, DHYLA, DHYLA, DHYLA, NN);
    k_softmax_row<<<NN, 256, 0, stream>>>(S, NN, 0.25f);
    k_kth_row<<<NN, 256, 0, stream>>>(S, NN, TOPK_, kth);
    n = NN * NN;       k_thresh_sym<<<(n + 255) / 256, 256, 0, stream>>>(S, kth, NN);

    // ---- degrees + mixed adjacencies ----
    k_rowsum<<<NN, 256, 0, stream>>>(S, deg_l, NN);
    k_rowsum<<<NN, 256, 0, stream>>>(A, deg_a, NN);
    k_deginv<<<(NN + 255) / 256, 256, 0, stream>>>(deg_l, dls, dlr, NN);
    k_deginv<<<(NN + 255) / 256, 256, 0, stream>>>(deg_a, das, dar, NN);
    n = NN * NN;       k_mix_adj<<<(n + 255) / 256, 256, 0, stream>>>(out_Asym, Asym_bf, S, A, dls, das, ratio, NN);

    // ---- structure (Ricci) loss: top-k of init_adj, gather A_row ----
    k_topk_idx<<<NN, 256, 0, stream>>>(init_adj, NN, TOPK_, tgt);
    n = NN * TOPK_;    k_structure<<<(n + 255) / 256, 256, 0, stream>>>(tgt, S, A, fvec, dlr, dar, ratio, accum, NN, TOPK_);

    // ---- 2-layer GCN (all B operands pre-transposed to NxK bf16) ----
    gemm32_wmma_bf16<<<512, 128, 0, stream>>>(feat_bf, w0t, XW, NN, DHID, DIN, DIN, DIN, DHID);
    k_transpose_f32_bf16<<<dim3(DHID / 32, NN / 32), dim3(32, 8), 0, stream>>>(XWt, XW, NN, DHID);
    gemm32_wmma_bf16<<<512, 128, 0, stream>>>(Asym_bf, XWt, hbuf, NN, DHID, NN, NN, NN, DHID);
    n = NN * DHID;     k_bias_ep<<<(n + 255) / 256, 256, 0, stream>>>(hbuf, NN, DHID, DHID, enc_b0, 1);
    k_f32_to_bf16<<<(n + 255) / 256, 256, 0, stream>>>(h_bf, hbuf, n);
    gemm32_wmma_bf16<<<256, 128, 0, stream>>>(h_bf, w1t, HW1, NN, DHYLA, DHID, DHID, DHID, DHYLA);
    k_transpose_f32_bf16<<<dim3(DHYLA / 32, NN / 32), dim3(32, 8), 0, stream>>>(HW1t, HW1, NN, DHYLA);
    gemm32_wmma_bf16<<<256, 128, 0, stream>>>(Asym_bf, HW1t, zE, NN, DHYLA, NN, NN, NN, DHYLA);
    n = NN * DHYLA;    k_bias_ep<<<(n + 255) / 256, 256, 0, stream>>>(zE, NN, DHYLA, DHYLA, enc_b1, 0);

    // ---- outputs: product_z, new_feature ----
    n = NN * 320;      k_copy_pz<<<(n + 255) / 256, 256, 0, stream>>>(out_pz, zE, emb_h, emb_s, NN);
    n = NN * 576;      k_copy_nf<<<(n + 255) / 256, 256, 0, stream>>>(out_nf, feature, emb_h, emb_s, NN);

    // ---- contrastive (InfoNCE) loss ----
    k_x12<<<(NN + 255) / 256, 256, 0, stream>>>(zE, x, x1_bf, x2_bf, nrm1, nrm2, NN);
    gemm32_wmma_bf16<<<4096, 128, 0, stream>>>(x1_bf, x2_bf, S, NN, NN, 512, 512, 512, NN);
    n = NN * NN;       k_sim_ep<<<(n + 255) / 256, 256, 0, stream>>>(S, nrm1, nrm2, NN);
    k_rowsum<<<NN, 256, 0, stream>>>(S, rowsum, NN);
    k_colsum<<<NN, 256, 0, stream>>>(S, colsum, NN);
    k_diag<<<(NN + 255) / 256, 256, 0, stream>>>(S, diagv, NN);
    k_info<<<1, 256, 0, stream>>>(rowsum, colsum, diagv, accum, NN);
    k_final<<<1, 1, 0, stream>>>(accum, out_loss, NN, TOPK_);
}
